// Basic_RT_45414984188518
// MI455X (gfx1250) — matrix operand order repaired and checked by reading, not compiled, not run on hardware
//
#include <hip/hip_runtime.h>

// ---------------------------------------------------------------------------
// MI455X (gfx1250, wave32) graph-transformer block.
// All GEMMs -> v_wmma_f32_16x16x32_bf16 (f32 accumulate).
// Fused kernels use 512 threads (16 waves), ONE 16-row m-strip per wave:
// no outer strip loop -> no loop-invariant B-fragment hoisting -> no spills.
// ---------------------------------------------------------------------------

typedef __attribute__((ext_vector_type(16))) __bf16 v16bf;
typedef __attribute__((ext_vector_type(8)))  __bf16 v8bf;
typedef __attribute__((ext_vector_type(8)))  float  v8f;

#define DEVINL __device__ __forceinline__
#define ALD 136   // LDS row stride (bf16/f32 elems): 16B-aligned, bank-rotating
#define TLD 264   // LDS row stride for ATT / V2^T (256 + 8)

DEVINL __bf16 f2bf(float x) {
  unsigned u = __builtin_bit_cast(unsigned, x);
  unsigned r = u + 0x7FFFu + ((u >> 16) & 1u);   // round-to-nearest-even
  unsigned short h = (unsigned short)(r >> 16);
  return __builtin_bit_cast(__bf16, h);
}
DEVINL float bf2f(__bf16 h) {
  unsigned short s = __builtin_bit_cast(unsigned short, h);
  unsigned u = ((unsigned)s) << 16;
  return __builtin_bit_cast(float, u);
}

DEVINL v16bf make16(v8bf lo, v8bf hi) {
  return __builtin_shufflevector(lo, hi, 0, 1, 2, 3, 4, 5, 6, 7,
                                 8, 9, 10, 11, 12, 13, 14, 15);
}

DEVINL v8f wmma_bf16(v16bf a, v16bf b, v8f c) {
  return __builtin_amdgcn_wmma_f32_16x16x32_bf16(false, a, false, b, (short)0, c,
                                                 false, false);
}

// A fragment (16x32) from row-major bf16 (global or LDS); two 16B loads.
DEVINL v16bf load_a(const __bf16* A, int lda, int m0, int k0, int lane) {
  const v8bf* p = (const v8bf*)(A + (size_t)(m0 + (lane & 15)) * lda +
                                k0 + ((lane >> 4) << 4));
  return make16(p[0], p[1]);
}

// B fragment from pre-packed weights: 32B contiguous per lane.
DEVINL v16bf load_b(const __bf16* P, int nct, int kt, int nt, int lane) {
  const v8bf* p = (const v8bf*)(P + ((size_t)(kt * nct + nt) * 32 + lane) * 16);
  return make16(p[0], p[1]);
}

// B fragment from an N-major (transposed) bf16 matrix S[n][k], row stride ld.
DEVINL v16bf load_bt(const __bf16* S, int ld, int k0, int n0, int lane) {
  const v8bf* p = (const v8bf*)(S + (size_t)(n0 + (lane & 15)) * ld +
                                k0 + ((lane >> 4) << 4));
  return make16(p[0], p[1]);
}

// ---------------------------------------------------------------------------
// prep kernels
// ---------------------------------------------------------------------------
__global__ void k_cvt8(const float* __restrict__ src, __bf16* __restrict__ dst,
                       int n8) {
  int i = blockIdx.x * blockDim.x + threadIdx.x;
  if (i >= n8) return;
  const float4* s = (const float4*)src + (size_t)i * 2;
  float4 a = s[0], b = s[1];
  v8bf o;
  o[0] = f2bf(a.x); o[1] = f2bf(a.y); o[2] = f2bf(a.z); o[3] = f2bf(a.w);
  o[4] = f2bf(b.x); o[5] = f2bf(b.y); o[6] = f2bf(b.z); o[7] = f2bf(b.w);
  ((v8bf*)dst)[i] = o;
}

__global__ void k_pack_b(const float* __restrict__ W, __bf16* __restrict__ P,
                         int K, int N) {
  int idx = blockIdx.x * blockDim.x + threadIdx.x;
  int ntiles = N >> 4;
  int total = (K >> 5) * ntiles * 32;
  if (idx >= total) return;
  int lane = idx & 31;
  int tile = idx >> 5;
  int nt = tile % ntiles, kt = tile / ntiles;
  int col = nt * 16 + (lane & 15);
  int ks = kt * 32 + ((lane >> 4) << 4);
  __bf16* dst = P + (size_t)idx * 16;
#pragma unroll
  for (int t = 0; t < 16; ++t) dst[t] = f2bf(W[(size_t)(ks + t) * N + col]);
}

__global__ void k_rowgemm(const float* __restrict__ x, const float* __restrict__ W,
                          const float* __restrict__ bias, float* __restrict__ y,
                          int K, int N) {
  int b = blockIdx.x, n = threadIdx.x;
  if (n >= N) return;
  float acc = bias[n];
  for (int k = 0; k < K; ++k) acc += x[b * K + k] * W[(size_t)k * N + n];
  y[b * N + n] = acc;
}

// ---------------------------------------------------------------------------
// generic WMMA GEMM: one wave per 16-row m-strip, 256 thr -> 128 rows/block.
// ---------------------------------------------------------------------------
template <int KSTEPS, int RELU, int RES, int OUTBF>
__global__ __launch_bounds__(256)
void k_gemm(const __bf16* __restrict__ A, const __bf16* __restrict__ P,
            const float* __restrict__ bias, const float* __restrict__ residual,
            float* __restrict__ Cf, __bf16* __restrict__ Cb, int N) {
  const int K = KSTEPS * 32;
  const int lane = threadIdx.x & 31, wave = threadIdx.x >> 5;
  const int m0 = blockIdx.x * 128 + wave * 16;
  const int nct = N >> 4;
  v16bf aF[KSTEPS];
#pragma unroll
  for (int ks = 0; ks < KSTEPS; ++ks) aF[ks] = load_a(A, K, m0, ks * 32, lane);
  for (int nt = 0; nt < nct; ++nt) {
    v8f acc = {};
#pragma unroll
    for (int ks = 0; ks < KSTEPS; ++ks)
      acc = wmma_bf16(aF[ks], load_b(P, nct, ks, nt, lane), acc);
    int col = nt * 16 + (lane & 15), half = lane >> 4;
    float bb = bias[col];
#pragma unroll
    for (int r = 0; r < 8; ++r) {
      int row = m0 + r + 8 * half;
      float v = acc[r] + bb;
      if (RES) v += residual[(size_t)row * N + col];
      if (RELU) v = fmaxf(v, 0.f);
      if (OUTBF) Cb[(size_t)row * N + col] = f2bf(v);
      else       Cf[(size_t)row * N + col] = v;
    }
  }
}

// ---------------------------------------------------------------------------
// LayerNorm rows of 128; one wave per row; optional secondary bf16 output.
// ---------------------------------------------------------------------------
__global__ void k_layernorm(const float* __restrict__ X, const float* __restrict__ s,
                            const float* __restrict__ o, float* __restrict__ Y,
                            __bf16* __restrict__ Yb, int rows) {
  int lane = threadIdx.x & 31, wave = threadIdx.x >> 5;
  int row = blockIdx.x * 8 + wave;
  if (row >= rows) return;
  const float* x = X + (size_t)row * 128;
  float v0 = x[lane], v1 = x[lane + 32], v2 = x[lane + 64], v3 = x[lane + 96];
  float sum = v0 + v1 + v2 + v3;
  float sq  = v0 * v0 + v1 * v1 + v2 * v2 + v3 * v3;
#pragma unroll
  for (int off = 16; off; off >>= 1) {
    sum += __shfl_xor(sum, off);
    sq  += __shfl_xor(sq, off);
  }
  float mean = sum * (1.f / 128.f);
  float inv  = rsqrtf(sq * (1.f / 128.f) - mean * mean + 1e-5f);
  float r0 = (v0 - mean) * inv * s[lane]      + o[lane];
  float r1 = (v1 - mean) * inv * s[lane + 32] + o[lane + 32];
  float r2 = (v2 - mean) * inv * s[lane + 64] + o[lane + 64];
  float r3 = (v3 - mean) * inv * s[lane + 96] + o[lane + 96];
  float* y = Y + (size_t)row * 128;
  y[lane] = r0; y[lane + 32] = r1; y[lane + 64] = r2; y[lane + 96] = r3;
  if (Yb) {
    __bf16* yb = Yb + (size_t)row * 128;
    yb[lane] = f2bf(r0); yb[lane + 32] = f2bf(r1);
    yb[lane + 64] = f2bf(r2); yb[lane + 96] = f2bf(r3);
  }
}

// ---------------------------------------------------------------------------
// Fused attention: one block (512 thr, 16 waves) per (b,i); 1 strip/wave.
// ---------------------------------------------------------------------------
__global__ __launch_bounds__(512)
void k_attention(const __bf16* __restrict__ edgeBf,
                 const __bf16* __restrict__ wq, const __bf16* __restrict__ wk,
                 const __bf16* __restrict__ wv,
                 const float* __restrict__ bq, const float* __restrict__ bk,
                 const float* __restrict__ bv,
                 const float* __restrict__ nQ, const float* __restrict__ nK,
                 const float* __restrict__ nV,
                 const float* __restrict__ gQ, const float* __restrict__ gK,
                 const float* __restrict__ gV,
                 __bf16* __restrict__ attwBf) {
  extern __shared__ char smem[];
  __bf16* sQ = (__bf16*)smem;            // 256*ALD; reused as V2^T (128*TLD)
  __bf16* sK = sQ + 256 * ALD;           // 256*ALD
  __bf16* sA = sK + 256 * ALD;           // 16*TLD  (ATT, bf16)
  const int lane = threadIdx.x & 31, wave = threadIdx.x >> 5;
  const int b = blockIdx.x >> 8, i = blockIdx.x & 255;
  const __bf16* E = edgeBf + (size_t)(b * 256 + i) * 256 * 128;

  // ---- phase 1: eQ, eK (A fragments loaded once per strip) ----
  {
    int m0 = wave * 16;
    v16bf aF[4];
#pragma unroll
    for (int ks = 0; ks < 4; ++ks) aF[ks] = load_a(E, 128, m0, ks * 32, lane);
#pragma unroll
    for (int which = 0; which < 2; ++which) {
      const __bf16* W    = which ? wk : wq;
      const float*  bias = which ? bk : bq;
      __bf16*       OUT  = which ? sK : sQ;
      for (int nt = 0; nt < 8; ++nt) {
        v8f acc = {};
#pragma unroll
        for (int ks = 0; ks < 4; ++ks)
          acc = wmma_bf16(aF[ks], load_b(W, 8, ks, nt, lane), acc);
        int col = nt * 16 + (lane & 15), half = lane >> 4;
        float bs = bias[col];
#pragma unroll
        for (int r = 0; r < 8; ++r)
          OUT[(m0 + r + 8 * half) * ALD + col] = f2bf(acc[r] + bs);
      }
    }
  }
  __syncthreads();

  // ---- phase 2: scores + softmax (waves 0..7, wave == head h) ----
  if (wave < 8) {
    int h = wave;
    float aq[16], gk[16];
#pragma unroll
    for (int d = 0; d < 16; ++d) {
      int c = h * 16 + d;
      aq[d] = nQ[(size_t)(b * 256 + i) * 128 + c] + gQ[b * 128 + c];
      gk[d] = gK[b * 128 + c];
    }
    float sc[8];
#pragma unroll
    for (int jj = 0; jj < 8; ++jj) {
      int j = lane + jj * 32;
      const v8bf* q8 = (const v8bf*)(sQ + j * ALD + h * 16);
      const v8bf* k8 = (const v8bf*)(sK + j * ALD + h * 16);
      v8bf q0 = q8[0], q1 = q8[1], k0 = k8[0], k1 = k8[1];
      const float4* nk4 = (const float4*)(nK + (size_t)(b * 256 + j) * 128 + h * 16);
      float4 n0 = nk4[0], n1 = nk4[1], n2 = nk4[2], n3 = nk4[3];
      float nk[16] = {n0.x, n0.y, n0.z, n0.w, n1.x, n1.y, n1.z, n1.w,
                      n2.x, n2.y, n2.z, n2.w, n3.x, n3.y, n3.z, n3.w};
      float s = 0.f;
#pragma unroll
      for (int d = 0; d < 16; ++d) {
        float q  = bf2f(d < 8 ? q0[d] : q1[d - 8]) + aq[d];
        float kk = bf2f(d < 8 ? k0[d] : k1[d - 8]) + nk[d] + gk[d];
        s += q * kk;
      }
      sc[jj] = s * 0.25f;   // 1/sqrt(HS), HS=16
    }
    float mx = sc[0];
#pragma unroll
    for (int jj = 1; jj < 8; ++jj) mx = fmaxf(mx, sc[jj]);
#pragma unroll
    for (int off = 16; off; off >>= 1) mx = fmaxf(mx, __shfl_xor(mx, off));
    float sum = 0.f;
#pragma unroll
    for (int jj = 0; jj < 8; ++jj) { sc[jj] = __expf(sc[jj] - mx); sum += sc[jj]; }
#pragma unroll
    for (int off = 16; off; off >>= 1) sum += __shfl_xor(sum, off);
    float inv = 1.f / sum;
#pragma unroll
    for (int jj = 0; jj < 8; ++jj)
      sA[h * TLD + lane + jj * 32] = f2bf(sc[jj] * inv);
    v8bf z = {};
    ((v8bf*)(sA + (h + 8) * TLD))[lane] = z;   // zero pad row h+8
  }
  __syncthreads();

  // ---- phase 3: V2^T = (E @ wev + bv + nV + gV)^T into sQ region ----
  {
    __bf16* sVt = sQ;                     // 128 rows x TLD
    int m0 = wave * 16;
    v16bf aF[4];
#pragma unroll
    for (int ks = 0; ks < 4; ++ks) aF[ks] = load_a(E, 128, m0, ks * 32, lane);
    for (int nt = 0; nt < 8; ++nt) {
      v8f acc = {};
#pragma unroll
      for (int ks = 0; ks < 4; ++ks)
        acc = wmma_bf16(aF[ks], load_b(wv, 8, ks, nt, lane), acc);
      int col = nt * 16 + (lane & 15), half = lane >> 4;
      float bs = bv[col] + gV[b * 128 + col];
#pragma unroll
      for (int r = 0; r < 8; ++r) {
        int j = m0 + r + 8 * half;
        float v = acc[r] + bs + nV[(size_t)(b * 256 + j) * 128 + col];
        sVt[(size_t)col * TLD + j] = f2bf(v);
      }
    }
  }
  __syncthreads();

  // ---- phase 4: OUT = ATT(16x256) @ V2(256x128); wave nt -> n-tile nt ----
  if (wave < 8) {
    const __bf16* sVt = sQ;
    int nt = wave;
    v8f acc = {};
#pragma unroll
    for (int ks = 0; ks < 8; ++ks)
      acc = wmma_bf16(load_a(sA, TLD, 0, ks * 32, lane),
                      load_bt(sVt, TLD, ks * 32, nt * 16, lane), acc);
    if (lane < 16)
      attwBf[(size_t)(b * 256 + i) * 128 + nt * 16 + lane] = f2bf(acc[nt]);
  }
}

// ---------------------------------------------------------------------------
// Fused edge update: one block (512 thr, 16 waves) per (b,i); 1 strip/wave.
// ---------------------------------------------------------------------------
DEVINL const __bf16* cat_ptr(const __bf16* edgeBf, const __bf16* nodesBf,
                             const __bf16* graphBf, int b, int i, int jp, int c) {
  if (c < 128) return edgeBf  + ((size_t)(b * 256 + i) * 256 + jp) * 128 + c;
  if (c < 256) return edgeBf  + ((size_t)(b * 256 + jp) * 256 + i) * 128 + (c - 128);
  if (c < 384) return nodesBf + (size_t)(b * 256 + jp) * 128 + (c - 256);
  if (c < 512) return nodesBf + (size_t)(b * 256 + i) * 128 + (c - 384);
  return graphBf + b * 128 + (c - 512);
}

DEVINL void ln_rows(const float* sT, const float* s, const float* o,
                    __bf16* dstLds, float* dstGlb, size_t glbBase,
                    int lane, int wave) {
  for (int row = wave; row < 256; row += 16) {
    const float* x = sT + row * ALD;
    float v0 = x[lane], v1 = x[lane + 32], v2 = x[lane + 64], v3 = x[lane + 96];
    float sum = v0 + v1 + v2 + v3;
    float sq  = v0 * v0 + v1 * v1 + v2 * v2 + v3 * v3;
#pragma unroll
    for (int off = 16; off; off >>= 1) {
      sum += __shfl_xor(sum, off);
      sq  += __shfl_xor(sq, off);
    }
    float mean = sum * (1.f / 128.f);
    float inv  = rsqrtf(sq * (1.f / 128.f) - mean * mean + 1e-5f);
    float r0 = (v0 - mean) * inv * s[lane]      + o[lane];
    float r1 = (v1 - mean) * inv * s[lane + 32] + o[lane + 32];
    float r2 = (v2 - mean) * inv * s[lane + 64] + o[lane + 64];
    float r3 = (v3 - mean) * inv * s[lane + 96] + o[lane + 96];
    if (dstLds) {
      __bf16* y = dstLds + row * ALD;
      y[lane] = f2bf(r0); y[lane + 32] = f2bf(r1);
      y[lane + 64] = f2bf(r2); y[lane + 96] = f2bf(r3);
    } else {
      float* y = dstGlb + glbBase + (size_t)row * 128;
      y[lane] = r0; y[lane + 32] = r1; y[lane + 64] = r2; y[lane + 96] = r3;
    }
  }
}

__global__ __launch_bounds__(512)
void k_edge(const __bf16* __restrict__ edgeBf,
            const float* __restrict__ edgeF,   // residual (f32)
            const __bf16* __restrict__ nodesBf,
            const __bf16* __restrict__ graphBf,
            const __bf16* __restrict__ w1, const float* __restrict__ b1,
            const __bf16* __restrict__ w2, const float* __restrict__ b2,
            const __bf16* __restrict__ w3, const float* __restrict__ b3,
            const __bf16* __restrict__ w4, const float* __restrict__ b4,
            const float* __restrict__ ln1s, const float* __restrict__ ln1o,
            const float* __restrict__ ln2s, const float* __restrict__ ln2o,
            float* __restrict__ out) {
  extern __shared__ char smem[];
  __bf16* sH  = (__bf16*)smem;             // 256*ALD bf16: h1 / h2
  float*  sT  = (float*)(sH + 256 * ALD);  // 256*ALD f32 : pre-LN activations
  __bf16* sE1 = (__bf16*)(sT + 256 * ALD); // 256*ALD bf16: e1 (post-LN1)
  const int lane = threadIdx.x & 31, wave = threadIdx.x >> 5;
  const int b = blockIdx.x >> 8, i = blockIdx.x & 255;
  const size_t ebase = (size_t)(b * 256 + i) * 256 * 128;
  const int m0 = wave * 16;

  // ---- stage 1: h1 = relu(cat @ el1 + b1); 8 accumulators, A loaded once ----
  {
    v8f acc[8] = {};
    for (int ks = 0; ks < 20; ++ks) {
      int row = m0 + (lane & 15);
      int kb  = ks * 32 + ((lane >> 4) << 4);
      const v8bf* lo = (const v8bf*)cat_ptr(edgeBf, nodesBf, graphBf, b, i, row, kb);
      const v8bf* hi = (const v8bf*)cat_ptr(edgeBf, nodesBf, graphBf, b, i, row, kb + 8);
      v16bf a = make16(lo[0], hi[0]);
#pragma unroll
      for (int nt = 0; nt < 8; ++nt)
        acc[nt] = wmma_bf16(a, load_b(w1, 8, ks, nt, lane), acc[nt]);
    }
#pragma unroll
    for (int nt = 0; nt < 8; ++nt) {
      int col = nt * 16 + (lane & 15), half = lane >> 4;
      float bs = b1[col];
#pragma unroll
      for (int r = 0; r < 8; ++r)
        sH[(m0 + r + 8 * half) * ALD + col] = f2bf(fmaxf(acc[nt][r] + bs, 0.f));
    }
  }
  __syncthreads();

  // ---- stage 2: t = edge + h1 @ el2 + b2 -> sT (f32) ----
  {
    v16bf aF[4];
#pragma unroll
    for (int ks = 0; ks < 4; ++ks) aF[ks] = load_a(sH, ALD, m0, ks * 32, lane);
    for (int nt = 0; nt < 8; ++nt) {
      v8f acc = {};
#pragma unroll
      for (int ks = 0; ks < 4; ++ks)
        acc = wmma_bf16(aF[ks], load_b(w2, 8, ks, nt, lane), acc);
      int col = nt * 16 + (lane & 15), half = lane >> 4;
      float bs = b2[col];
#pragma unroll
      for (int r = 0; r < 8; ++r) {
        int row = m0 + r + 8 * half;
        sT[row * ALD + col] = acc[r] + bs + edgeF[ebase + (size_t)row * 128 + col];
      }
    }
  }
  __syncthreads();

  // ---- stage 3: e1 = LN(t) -> sE1 ----
  ln_rows(sT, ln1s, ln1o, sE1, nullptr, 0, lane, wave);
  __syncthreads();

  // ---- stage 4: h2 = relu(e1 @ el3 + b3) -> sH ----
  {
    v16bf aF[4];
#pragma unroll
    for (int ks = 0; ks < 4; ++ks) aF[ks] = load_a(sE1, ALD, m0, ks * 32, lane);
    for (int nt = 0; nt < 8; ++nt) {
      v8f acc = {};
#pragma unroll
      for (int ks = 0; ks < 4; ++ks)
        acc = wmma_bf16(aF[ks], load_b(w3, 8, ks, nt, lane), acc);
      int col = nt * 16 + (lane & 15), half = lane >> 4;
      float bs = b3[col];
#pragma unroll
      for (int r = 0; r < 8; ++r)
        sH[(m0 + r + 8 * half) * ALD + col] = f2bf(fmaxf(acc[r] + bs, 0.f));
    }
  }
  __syncthreads();

  // ---- stage 5: t2 = e1 + h2 @ el4 + b4 -> sT ----
  {
    v16bf aF[4];
#pragma unroll
    for (int ks = 0; ks < 4; ++ks) aF[ks] = load_a(sH, ALD, m0, ks * 32, lane);
    for (int nt = 0; nt < 8; ++nt) {
      v8f acc = {};
#pragma unroll
      for (int ks = 0; ks < 4; ++ks)
        acc = wmma_bf16(aF[ks], load_b(w4, 8, ks, nt, lane), acc);
      int col = nt * 16 + (lane & 15), half = lane >> 4;
      float bs = b4[col];
#pragma unroll
      for (int r = 0; r < 8; ++r) {
        int row = m0 + r + 8 * half;
        sT[row * ALD + col] = acc[r] + bs + bf2f(sE1[row * ALD + col]);
      }
    }
  }
  __syncthreads();

  // ---- stage 6: edges_out = LN(t2) ----
  ln_rows(sT, ln2s, ln2o, nullptr, out, ebase, lane, wave);
}

// ---------------------------------------------------------------------------
// host orchestration
// ---------------------------------------------------------------------------
extern "C" void kernel_launch(void* const* d_in, const int* in_sizes, int n_in,
                              void* d_out, int out_size, void* d_ws, size_t ws_size,
                              hipStream_t stream) {
  const float* node  = (const float*)d_in[0];
  const float* edge  = (const float*)d_in[1];
  const float* graph = (const float*)d_in[2];
  auto F = [&](int idx) { return (const float*)d_in[idx]; };

  char* ws = (char*)d_ws;
  size_t off = 0;
  auto allocB = [&](size_t e) { __bf16* p = (__bf16*)(ws + off); off += e * 2; return p; };
  __bf16* Pnq = allocB(128 * 128); __bf16* Pnk = allocB(128 * 128); __bf16* Pnv = allocB(128 * 128);
  __bf16* Peq = allocB(128 * 128); __bf16* Pek = allocB(128 * 128); __bf16* Pev = allocB(128 * 128);
  __bf16* Pl1 = allocB(128 * 128); __bf16* Pl2 = allocB(128 * 256); __bf16* Pl3 = allocB(256 * 128);
  __bf16* Pe1 = allocB(640 * 128); __bf16* Pe2 = allocB(128 * 128);
  __bf16* Pe3 = allocB(128 * 128); __bf16* Pe4 = allocB(128 * 128);
  __bf16* edgeBf  = allocB((size_t)4 * 256 * 256 * 128);
  __bf16* nodeBf  = allocB(1024 * 128);
  __bf16* nodesBf = allocB(1024 * 128);
  __bf16* graphBf = allocB(4 * 128);
  __bf16* attwBf  = allocB(1024 * 128);
  __bf16* n1Bf    = allocB(1024 * 128);
  __bf16* t2Bf    = allocB(1024 * 256);
  off = (off + 255) & ~(size_t)255;
  auto allocF = [&](size_t e) { float* p = (float*)(ws + off); off += e * 4; return p; };
  float* fnQ = allocF(1024 * 128); float* fnK = allocF(1024 * 128); float* fnV = allocF(1024 * 128);
  float* fgQ = allocF(4 * 128);    float* fgK = allocF(4 * 128);    float* fgV = allocF(4 * 128);
  float* fT1 = allocF(1024 * 128);
  float* fN1 = allocF(1024 * 128);
  float* fT3 = allocF(1024 * 128);

  auto cvt = [&](const float* s, __bf16* d, size_t n) {
    int n8 = (int)(n / 8);
    k_cvt8<<<dim3((n8 + 255) / 256), dim3(256), 0, stream>>>(s, d, n8);
  };
  cvt(edge, edgeBf, (size_t)4 * 256 * 256 * 128);
  cvt(node, nodeBf, 1024 * 128);
  cvt(graph, graphBf, 4 * 128);
  auto pack = [&](int idx, __bf16* dst, int K, int N) {
    int total = (K >> 5) * (N >> 4) * 32;
    k_pack_b<<<dim3((total + 255) / 256), dim3(256), 0, stream>>>(F(idx), dst, K, N);
  };
  pack(5, Pnq, 128, 128);  pack(7, Pnk, 128, 128);  pack(9, Pnv, 128, 128);
  pack(11, Peq, 128, 128); pack(13, Pek, 128, 128); pack(15, Pev, 128, 128);
  pack(23, Pl1, 128, 128); pack(25, Pl2, 128, 256); pack(27, Pl3, 256, 128);
  pack(33, Pe1, 640, 128); pack(35, Pe2, 128, 128); pack(37, Pe3, 128, 128);
  pack(39, Pe4, 128, 128);

  k_gemm<4, 0, 0, 0><<<dim3(8), dim3(256), 0, stream>>>(nodeBf, Pnq, F(6),  nullptr, fnQ, nullptr, 128);
  k_gemm<4, 0, 0, 0><<<dim3(8), dim3(256), 0, stream>>>(nodeBf, Pnk, F(8),  nullptr, fnK, nullptr, 128);
  k_gemm<4, 0, 0, 0><<<dim3(8), dim3(256), 0, stream>>>(nodeBf, Pnv, F(10), nullptr, fnV, nullptr, 128);
  k_rowgemm<<<dim3(4), dim3(128), 0, stream>>>(graph, F(17), F(18), fgQ, 128, 128);
  k_rowgemm<<<dim3(4), dim3(128), 0, stream>>>(graph, F(19), F(20), fgK, 128, 128);
  k_rowgemm<<<dim3(4), dim3(128), 0, stream>>>(graph, F(21), F(22), fgV, 128, 128);

  size_t attSmem = (size_t)2 * 256 * ALD * 2 + (size_t)16 * TLD * 2;
  k_attention<<<dim3(1024), dim3(512), attSmem, stream>>>(
      edgeBf, Peq, Pek, Pev, F(12), F(14), F(16),
      fnQ, fnK, fnV, fgQ, fgK, fgV, attwBf);

  float* outNodes = (float*)d_out;
  float* outEdges = outNodes + 1024 * 128;
  k_gemm<4, 0, 1, 0><<<dim3(8), dim3(256), 0, stream>>>(attwBf, Pl1, F(24), node, fT1, nullptr, 128);
  k_layernorm<<<dim3(128), dim3(256), 0, stream>>>(fT1, F(29), F(30), fN1, n1Bf, 1024);
  k_gemm<4, 1, 0, 1><<<dim3(8), dim3(256), 0, stream>>>(n1Bf, Pl2, F(26), nullptr, nullptr, t2Bf, 256);
  k_gemm<8, 0, 1, 0><<<dim3(8), dim3(256), 0, stream>>>(t2Bf, Pl3, F(28), fN1, fT3, nullptr, 128);
  k_layernorm<<<dim3(128), dim3(256), 0, stream>>>(fT3, F(31), F(32), outNodes, nodesBf, 1024);

  size_t edgeSmem = (size_t)256 * ALD * (2 + 4 + 2);
  k_edge<<<dim3(1024), dim3(512), edgeSmem, stream>>>(
      edgeBf, edge, nodesBf, graphBf,
      Pe1, F(34), Pe2, F(36), Pe3, F(38), Pe4, F(40),
      F(41), F(42), F(43), F(44), outEdges);
}
